// BottleneckBlock_11793980194930
// MI455X (gfx1250) — compile-verified, matrix operand order checked by reading, not run on hardware
//
#include <hip/hip_runtime.h>
#include <hip/hip_bf16.h>
#include <stdint.h>

// Problem constants (match reference)
#define BDIM  4
#define NNODE 49152
#define CDIM  256
#define EDGE  393216
#define CMIDD 64
#define MROWS (BDIM * NNODE)      // 196608
#define EPSV  1e-5f

// WMMA vector types (gfx1250, wave32)
typedef __attribute__((ext_vector_type(16))) __bf16 v16bf;
typedef __attribute__((ext_vector_type(8)))  float  v8f;

__device__ __forceinline__ unsigned short f2bf_bits(float f) {
    union { float f; unsigned u; } a; a.f = f;
    unsigned r = a.u + 0x7FFFu + ((a.u >> 16) & 1u);   // round-to-nearest-even
    return (unsigned short)(r >> 16);
}
__device__ __forceinline__ unsigned pack2bf(float lo, float hi) {
    return (unsigned)f2bf_bits(lo) | ((unsigned)f2bf_bits(hi) << 16);
}

// ---------------------------------------------------------------------------
// Elementwise fill: dst[i] = scale*src[i]  (src==nullptr -> 0)
// ---------------------------------------------------------------------------
__global__ void scale_fill_kernel(float* __restrict__ dst, const float* __restrict__ src,
                                  float scale, size_t n) {
    size_t i = (size_t)blockIdx.x * blockDim.x + threadIdx.x;
    size_t stride = (size_t)gridDim.x * blockDim.x;
    if (src) {
        for (; i < n; i += stride) dst[i] = scale * src[i];
    } else {
        for (; i < n; i += stride) dst[i] = 0.0f;
    }
}

// ---------------------------------------------------------------------------
// Degree / normalized adjacency
// ---------------------------------------------------------------------------
__global__ void degree_kernel(const int* __restrict__ rows, const float* __restrict__ w,
                              float* __restrict__ deg) {
    int e = blockIdx.x * blockDim.x + threadIdx.x;
    if (e < EDGE) atomicAdd(&deg[rows[e]], w[e]);
}

__global__ void dinv_kernel(float* __restrict__ deg) {
    int i = blockIdx.x * blockDim.x + threadIdx.x;
    if (i < NNODE) {
        float d = deg[i];
        deg[i] = (d > 0.0f) ? rsqrtf(fmaxf(d, EPSV)) : 0.0f;
    }
}

__global__ void anorm_kernel(const int* __restrict__ rows, const int* __restrict__ cols,
                             const float* __restrict__ w, const float* __restrict__ dinv,
                             float* __restrict__ an) {
    int e = blockIdx.x * blockDim.x + threadIdx.x;
    if (e < EDGE) an[e] = w[e] * dinv[rows[e]] * dinv[cols[e]];
}

// ---------------------------------------------------------------------------
// Chebyshev propagation:  Xout[:, r, f] += coef * a_norm[e] * Xin[:, c, f]
// (Tx1: Xout pre-zeroed, coef = -1.  Tx2: Xout pre-set to -Tx0, coef = -2.)
// One thread handles one edge x 4 channels x all 4 batches.
// ---------------------------------------------------------------------------
__global__ void cheb_prop_kernel(const float* __restrict__ Xin, float* __restrict__ Xout,
                                 const int* __restrict__ rows, const int* __restrict__ cols,
                                 const float* __restrict__ an, int F, float coef) {
    const int chunks = F >> 2;
    long long idx = (long long)blockIdx.x * blockDim.x + threadIdx.x;
    long long tot = (long long)EDGE * chunks;
    if (idx >= tot) return;
    int e = (int)(idx / chunks);
    int f = (int)(idx % chunks) << 2;
    int r = rows[e], c = cols[e];
    float a = coef * an[e];
#pragma unroll
    for (int b = 0; b < BDIM; ++b) {
        const float4 v = *(const float4*)&Xin[((size_t)b * NNODE + c) * F + f];
        float* dst = &Xout[((size_t)b * NNODE + r) * F + f];
        atomicAdd(dst + 0, a * v.x);
        atomicAdd(dst + 1, a * v.y);
        atomicAdd(dst + 2, a * v.z);
        atomicAdd(dst + 3, a * v.w);
    }
}

// ---------------------------------------------------------------------------
// Fused 3-term Chebyshev GEMM, bf16 WMMA with f32 accumulate:
//   Y[M,Nout] = sum_{t=0..2} Xt[M,Kin] * W[t][Kin,Nout]  + bias
// Accumulators live in VGPRs across all 3 terms; Y written exactly once.
// Block: 256 thr = 8 waves; block tile 64(M) x 64(N); wave tile 16(M) x 32(N)
// (two f32 accumulators, A-fragment reused across both N-subtiles);
// K step 64 -> 4 v_wmma per barrier pair. A/B^T staged in LDS; fragments
// gathered as 32-bit LDS loads per the 16-bit A-matrix 16x32 VGPR layout.
// ---------------------------------------------------------------------------
#define BLK_M 64
#define BLK_N 64
#define BLK_K 64
#define LDA_S 68   // BLK_K + 4 pad (even, keeps 32-bit LDS alignment)
#define LDB_S 68

__global__ __launch_bounds__(256) void gemm3_bf16_wmma_kernel(
    const float* __restrict__ X0, const float* __restrict__ X1,
    const float* __restrict__ X2, const float* __restrict__ Wbase,
    const float* __restrict__ bias, float* __restrict__ Y,
    int Kin, int Nout) {
    __shared__ __bf16 As[BLK_M * LDA_S];
    __shared__ __bf16 Bs[BLK_N * LDB_S];   // transposed: Bs[n][k]

    const int tid  = threadIdx.x;
    const int lane = tid & 31;
    const int wave = tid >> 5;
    const int wm = wave & 3;   // 4 wave tiles along M
    const int wn = wave >> 2;  // 2 wave tiles along N (each covers 32 cols)
    const int m0 = blockIdx.x * BLK_M;
    const int n0 = blockIdx.y * BLK_N;
    const int lanemod = lane & 15;
    const int kg = lane >> 4;  // K-group half of the wave

    const float* Xs[3] = {X0, X1, X2};

    v8f acc0 = {};
    v8f acc1 = {};

#pragma unroll
    for (int t = 0; t < 3; ++t) {
        const float* __restrict__ X  = Xs[t];
        const float* __restrict__ Wm = Wbase + (size_t)t * Kin * Nout;

        for (int k0 = 0; k0 < Kin; k0 += BLK_K) {
            // Stage A tile (64 x 64 f32 -> bf16): float2 global loads,
            // packed 32-bit LDS stores.
            {
                int c2 = tid & 31;          // column pair 0..31 -> cols 2c2,2c2+1
                int r  = tid >> 5;          // 0..7
#pragma unroll
                for (int i = 0; i < 8; ++i) {
                    int rr = r + i * 8;
                    const float2 v =
                        *(const float2*)&X[(size_t)(m0 + rr) * Kin + (k0 + 2 * c2)];
                    *(unsigned*)&As[rr * LDA_S + 2 * c2] = pack2bf(v.x, v.y);
                }
                // prefetch next A tile (lowers to global_prefetch_b8)
                if (k0 + BLK_K < Kin)
                    __builtin_prefetch(&X[(size_t)(m0 + r) * Kin + (k0 + BLK_K)], 0, 0);
            }
            // Stage B tile transposed: Bs[n][k] = W[(k0+k)*Nout + n0+n],
            // paired along k for packed 32-bit LDS stores.
            {
                int n  = tid & 63;          // 0..63
                int kq = tid >> 6;          // 0..3 (pair index)
#pragma unroll
                for (int i = 0; i < 8; ++i) {
                    int kk = 2 * (kq + 4 * i);   // even k in 0..62
                    float wlo = Wm[(size_t)(k0 + kk) * Nout + (n0 + n)];
                    float whi = Wm[(size_t)(k0 + kk + 1) * Nout + (n0 + n)];
                    *(unsigned*)&Bs[n * LDB_S + kk] = pack2bf(wlo, whi);
                }
            }
            __syncthreads();

            const int rowA  = wm * 16 + lanemod;
            const int colB0 = wn * 32 + lanemod;
            const int colB1 = colB0 + 16;
#pragma unroll
            for (int h = 0; h < 2; ++h) {   // two K=32 slices
                union { v16bf v; unsigned u[8]; } afrag, bfrag0, bfrag1;
#pragma unroll
                for (int v = 0; v < 8; ++v) {
                    int kb = h * 32 +
                             ((v < 4) ? (kg * 8 + 2 * v) : (16 + kg * 8 + 2 * (v - 4)));
                    afrag.u[v]  = *(const unsigned*)&As[rowA  * LDA_S + kb];
                    bfrag0.u[v] = *(const unsigned*)&Bs[colB0 * LDB_S + kb];
                    bfrag1.u[v] = *(const unsigned*)&Bs[colB1 * LDB_S + kb];
                }
                // A-fragment reused for both N-subtiles.
                acc0 = __builtin_amdgcn_wmma_f32_16x16x32_bf16(
                    false, afrag.v, false, bfrag0.v, (short)0, acc0, false, false);
                acc1 = __builtin_amdgcn_wmma_f32_16x16x32_bf16(
                    false, afrag.v, false, bfrag1.v, (short)0, acc1, false, false);
            }
            __syncthreads();
        }
    }

    // Store D once: lane n = lane%16, VGPR v -> row v + 8*(lane/16)
    const int ncol0 = n0 + wn * 32 + lanemod;
    const int ncol1 = ncol0 + 16;
    const float bv0 = bias[ncol0];
    const float bv1 = bias[ncol1];
#pragma unroll
    for (int v = 0; v < 8; ++v) {
        int mrow = m0 + wm * 16 + v + 8 * kg;
        Y[(size_t)mrow * Nout + ncol0] = acc0[v] + bv0;
        Y[(size_t)mrow * Nout + ncol1] = acc1[v] + bv1;
    }
}

// ---------------------------------------------------------------------------
// InstanceNorm over node axis (per (b,c)) + ReLU, in-place; optional residual.
// ---------------------------------------------------------------------------
__global__ __launch_bounds__(256) void inst_norm_relu_kernel(
    float* __restrict__ Y, const float* __restrict__ residual, int F) {
    __shared__ float ssum[256];
    __shared__ float ssq[256];
    const int c = blockIdx.x;
    const int b = blockIdx.y;
    float* base = Y + (size_t)b * NNODE * F + c;
    float s = 0.0f, sq = 0.0f;
    for (int n = threadIdx.x; n < NNODE; n += 256) {
        float v = base[(size_t)n * F];
        s += v; sq += v * v;
    }
    ssum[threadIdx.x] = s; ssq[threadIdx.x] = sq;
    __syncthreads();
    for (int off = 128; off > 0; off >>= 1) {
        if (threadIdx.x < off) {
            ssum[threadIdx.x] += ssum[threadIdx.x + off];
            ssq[threadIdx.x]  += ssq[threadIdx.x + off];
        }
        __syncthreads();
    }
    const float mean = ssum[0] * (1.0f / NNODE);
    const float var  = ssq[0] * (1.0f / NNODE) - mean * mean;
    const float rstd = rsqrtf(var + EPSV);
    const float* rbase = residual ? residual + (size_t)b * NNODE * F + c : nullptr;
    for (int n = threadIdx.x; n < NNODE; n += 256) {
        float v = (base[(size_t)n * F] - mean) * rstd;
        v = fmaxf(v, 0.0f);
        if (rbase) v += rbase[(size_t)n * F];
        base[(size_t)n * F] = v;
    }
}

// ---------------------------------------------------------------------------
// Host orchestration
// ---------------------------------------------------------------------------
extern "C" void kernel_launch(void* const* d_in, const int* in_sizes, int n_in,
                              void* d_out, int out_size, void* d_ws, size_t ws_size,
                              hipStream_t stream) {
    const float* x   = (const float*)d_in[0];
    const int*   ei  = (const int*)d_in[1];
    const float* ew  = (const float*)d_in[2];
    const float* W1  = (const float*)d_in[3];
    const float* b1  = (const float*)d_in[4];
    const float* W2  = (const float*)d_in[5];
    const float* b2  = (const float*)d_in[6];
    const float* W3  = (const float*)d_in[7];
    const float* b3  = (const float*)d_in[8];
    float* out = (float*)d_out;

    const int* rows = ei;
    const int* cols = ei + EDGE;

    // Workspace carve (floats)
    float* ws   = (float*)d_ws;
    float* deg  = ws;                                   // NNODE
    float* an   = deg + NNODE;                          // EDGE
    float* bufA = an + EDGE;                            // MROWS*CDIM
    float* bufB = bufA + (size_t)MROWS * CDIM;          // MROWS*CDIM
    float* y1   = bufB + (size_t)MROWS * CDIM;          // MROWS*CMIDD
    // layer-2/3 reuse inside bufA/bufB
    float* t1   = bufA;
    float* y2   = bufA + (size_t)MROWS * CMIDD;
    float* t2   = bufB;
    float* t1b  = bufB + (size_t)MROWS * CMIDD;
    float* t2b  = bufB + 2 * (size_t)MROWS * CMIDD;

    // --- normalized adjacency ---
    scale_fill_kernel<<<256, 256, 0, stream>>>(deg, nullptr, 0.0f, (size_t)NNODE);
    degree_kernel<<<(EDGE + 255) / 256, 256, 0, stream>>>(rows, ew, deg);
    dinv_kernel<<<(NNODE + 255) / 256, 256, 0, stream>>>(deg);
    anorm_kernel<<<(EDGE + 255) / 256, 256, 0, stream>>>(rows, cols, ew, deg, an);

    auto run_layer = [&](const float* tx0, float* T1, float* T2,
                         const float* Wp, const float* bp,
                         int Kin, int Nout, float* Yout) {
        const size_t sz = (size_t)MROWS * Kin;
        const long long ptot = (long long)EDGE * (Kin >> 2);
        const int pblocks = (int)((ptot + 255) / 256);
        // Tx1 = -scatter(a * tx0[col])
        scale_fill_kernel<<<4096, 256, 0, stream>>>(T1, nullptr, 0.0f, sz);
        cheb_prop_kernel<<<pblocks, 256, 0, stream>>>(tx0, T1, rows, cols, an, Kin, -1.0f);
        // Tx2 = -2*scatter(a * Tx1[col]) - Tx0
        scale_fill_kernel<<<4096, 256, 0, stream>>>(T2, tx0, -1.0f, sz);
        cheb_prop_kernel<<<pblocks, 256, 0, stream>>>(T1, T2, rows, cols, an, Kin, -2.0f);
        // Y = Tx0*W[0] + Tx1*W[1] + Tx2*W[2] + b   (single fused GEMM)
        dim3 g(MROWS / BLK_M, Nout / BLK_N);
        gemm3_bf16_wmma_kernel<<<g, 256, 0, stream>>>(tx0, T1, T2, Wp, bp, Yout,
                                                      Kin, Nout);
    };

    // Layer 1: C=256 -> 64
    run_layer(x, bufA, bufB, W1, b1, CDIM, CMIDD, y1);
    inst_norm_relu_kernel<<<dim3(CMIDD, BDIM), 256, 0, stream>>>(y1, nullptr, CMIDD);

    // Layer 2: 64 -> 64
    run_layer(y1, t1, t2, W2, b2, CMIDD, CMIDD, y2);
    inst_norm_relu_kernel<<<dim3(CMIDD, BDIM), 256, 0, stream>>>(y2, nullptr, CMIDD);

    // Layer 3: 64 -> 256, output + residual
    run_layer(y2, t1b, t2b, W3, b3, CMIDD, CDIM, out);
    inst_norm_relu_kernel<<<dim3(CDIM, BDIM), 256, 0, stream>>>(out, x, CDIM);
}
